// STBlock_77996606096122
// MI455X (gfx1250) — matrix operand order repaired and checked by reading, not compiled, not run on hardware
//
#include <hip/hip_runtime.h>
#include <hip/hip_bf16.h>

typedef __attribute__((ext_vector_type(16))) _Float16 v16h;
typedef __attribute__((ext_vector_type(8)))  _Float16 v8h;
typedef __attribute__((ext_vector_type(4)))  _Float16 v4h;
typedef __attribute__((ext_vector_type(8)))  float    v8f;

namespace {
constexpr int B_  = 16;
constexpr int N_  = 307;
constexpr int T_  = 12;
constexpr int C_  = 64;
constexpr int H_  = 4;
constexpr int D3_ = 192;
constexpr int NT_ = N_ * T_;                     // 3684
constexpr int R_  = B_ * N_ * T_;                // 58944 rows (multiple of 64)
constexpr int TC_ = T_ * C_;                     // 768
constexpr long long AROW_ = (long long)N_ * TC_; // 235776 (one batch of x as [307,768])
constexpr int AN_ = N_ * N_;                     // 94249
constexpr long long SZ_  = (long long)R_ * C_;   // 3,772,416
constexpr long long SZ3_ = (long long)R_ * D3_;  // 11,317,248
constexpr long long MIDN_ = (long long)R_ * 256;

// -------- input leaf indices (setup_inputs dict insertion order, recursive) --------
enum {
  IN_X = 0, IN_ADJ = 1,
  A1_WQ = 2, A1_WK, A1_WV, A1_OW, A1_OB,          // agcn.att1
  AT2_WQ, AT2_WK, AT2_WV, AT2_OW, AT2_OB,         // agcn.att2 (7..11)
  FF_L1W, FF_L1B, FF_L2W, FF_L2B,                 // agcn.ff   (12..15)
  F1N_L1W, F1N_L1B, F1N_L2W, F1N_L2B,             // agcn.ff1n (16..19)
  FF1_W, FF1_B,                                   // agcn.ff1  (20,21)
  AFS_W, AFS_B, AFG_W, AFG_B,                     // agcn.fs/fg (22..25)
  E1_W, E1_B, E2_W, E2_B,                         // (26..29)
  N1_G, N1_B, N2_G, N2_B, N3_G, N3_B,             // agcn.n1..n6 (30..41)
  N4_G, N4_B, N5_G, N5_B, N6_G, N6_B,
  NORM1_G, NORM1_B, NORM2_G, NORM2_B,             // (42..49)
  NORM3_G, NORM3_B, NORM4_G, NORM4_B,
  ADP,                                            // (50)
  C4_W, C4_B, C5_W, C5_B,                         // (51..54)
  GG1_W, GG1_B, GG2_W, GG2_B,                     // (55..58)
  LG1_W, LG1_B, LG2_W, LG2_B,                     // (59..62)
  TG_WQ, TG_WK, TG_WV, TG_OW, TG_OB,              // (63..67)
  TL_WQ, TL_WK, TL_WV, TL_OW, TL_OB,              // (68..72)
  FS_W, FS_B, FG_W, FG_B,                         // (73..76)
  SA_WQ, SA_WK, SA_WV, SA_OW, SA_OB,              // stga.att (77..81)
  SF_L1W, SF_L1B, SF_L2W, SF_L2B,                 // stga.ffn (82..85)
  SL1_G, SL1_B, SL2_G, SL2_B,                     // stga.ln1/ln2 (86..89)
  OP_W, OP_B                                      // out_proj (90,91)
};
constexpr int LDA_ = 40;   // padded LDS row stride in halfs (80 B: keeps 16B alignment)
} // namespace

__device__ __forceinline__ float sigm(float v) { return 1.0f / (1.0f + __expf(-v)); }

// =====================================================================
// Generic WMMA GEMM:  C[M,Nn] = act(A[M,K] @ Wt[K,Nn] + bias)
// block = 256 thr (8 wave32), block tile 64x64, BK=32.
// Each wave computes a 16x32 strip -> 2x v_wmma_f32_16x16x32_f16 / K-step.
// f16 staging in LDS (A row-major, W transposed), f32 accumulate.
// Nn must be a multiple of 64.
// =====================================================================
template <int ACT>
__global__ __launch_bounds__(256) void gemm_wmma(
    const float* __restrict__ A, const float* __restrict__ Wt,
    const float* __restrict__ bias, float* __restrict__ C,
    int M, int Nn, int K, long long sA, long long sW, long long sC) {
  __shared__ _Float16 As[64 * LDA_];   // [row 0..63][k 0..31]
  __shared__ _Float16 BsT[64 * LDA_];  // [col 0..63][k 0..31]
  A  += (long long)blockIdx.z * sA;
  Wt += (long long)blockIdx.z * sW;
  C  += (long long)blockIdx.z * sC;
  const int rowBase = blockIdx.y * 64;
  const int colBase = blockIdx.x * 64;
  const int tid  = threadIdx.x;
  const int wave = tid >> 5;
  const int lane = tid & 31;
  const int hl = lane & 15, hi = lane >> 4;
  const int tm = (wave & 3) << 4;   // 0,16,32,48
  const int tn = (wave >> 2) << 5;  // 0,32
  v8f acc0 = {};
  v8f acc1 = {};
  const bool fullM = (rowBase + 64 <= M);
  const int kFull = fullM ? (K & ~31) : 0;
  int k0 = 0;
  // -------- branch-free full-tile loop (dominant path) --------
  for (; k0 < kFull; k0 += 32) {
#pragma unroll
    for (int u = 0; u < 2; ++u) {
      int fi = tid + (u << 8);           // 512 float4 tiles of A (64x32)
      int r = fi >> 3, c4 = (fi & 7) << 2;
      const float4 va = *(const float4*)&A[(long long)(rowBase + r) * K + k0 + c4];
      v4h ha; ha[0] = (_Float16)va.x; ha[1] = (_Float16)va.y;
      ha[2] = (_Float16)va.z; ha[3] = (_Float16)va.w;
      *(v4h*)&As[r * LDA_ + c4] = ha;
    }
#pragma unroll
    for (int u = 0; u < 2; ++u) {
      int fi = tid + (u << 8);           // 512 float4 tiles of W (32x64)
      int r = fi >> 4, c4 = (fi & 15) << 2;
      const float4 vb = *(const float4*)&Wt[(long long)(k0 + r) * Nn + colBase + c4];
      BsT[(c4 + 0) * LDA_ + r] = (_Float16)vb.x;
      BsT[(c4 + 1) * LDA_ + r] = (_Float16)vb.y;
      BsT[(c4 + 2) * LDA_ + r] = (_Float16)vb.z;
      BsT[(c4 + 3) * LDA_ + r] = (_Float16)vb.w;
    }
    if (k0 + 32 < K)
      __builtin_prefetch(&A[(long long)(rowBase + (tid >> 2)) * K + k0 + 32], 0, 1);
    __syncthreads();
    const v8h a0 = *(const v8h*)&As[(tm + hl) * LDA_ + (hi << 3)];
    const v8h a1 = *(const v8h*)&As[(tm + hl) * LDA_ + (hi << 3) + 16];
    const v16h af = __builtin_shufflevector(a0, a1, 0, 1, 2, 3, 4, 5, 6, 7,
                                            8, 9, 10, 11, 12, 13, 14, 15);
    const v8h b0 = *(const v8h*)&BsT[(tn + hl) * LDA_ + (hi << 3)];
    const v8h b1 = *(const v8h*)&BsT[(tn + hl) * LDA_ + (hi << 3) + 16];
    const v16h bf0 = __builtin_shufflevector(b0, b1, 0, 1, 2, 3, 4, 5, 6, 7,
                                             8, 9, 10, 11, 12, 13, 14, 15);
    const v8h b2 = *(const v8h*)&BsT[(tn + 16 + hl) * LDA_ + (hi << 3)];
    const v8h b3 = *(const v8h*)&BsT[(tn + 16 + hl) * LDA_ + (hi << 3) + 16];
    const v16h bf1 = __builtin_shufflevector(b2, b3, 0, 1, 2, 3, 4, 5, 6, 7,
                                             8, 9, 10, 11, 12, 13, 14, 15);
    acc0 = __builtin_amdgcn_wmma_f32_16x16x32_f16(false, af, false, bf0,
                                                  (short)0, acc0, false, false);
    acc1 = __builtin_amdgcn_wmma_f32_16x16x32_f16(false, af, false, bf1,
                                                  (short)0, acc1, false, false);
    __syncthreads();
  }
  // -------- guarded tail (ragged M or K): branchless clamp + select --------
  for (; k0 < K; k0 += 32) {
    for (int i = tid; i < 64 * 32; i += 256) {
      int r = i >> 5, c = i & 31;
      int gr = rowBase + r, gc = k0 + c;
      int grc = min(gr, M - 1), gcc = min(gc, K - 1);
      float v = A[(long long)grc * K + gcc];
      As[r * LDA_ + c] = (gr < M && gc < K) ? (_Float16)v : (_Float16)0.0f;
    }
    for (int i = tid; i < 32 * 64; i += 256) {
      int r = i >> 6, c = i & 63;
      int gk = k0 + r;
      int gkc = min(gk, K - 1);
      float v = Wt[(long long)gkc * Nn + colBase + c];
      BsT[c * LDA_ + r] = (gk < K) ? (_Float16)v : (_Float16)0.0f;
    }
    __syncthreads();
    const v8h a0 = *(const v8h*)&As[(tm + hl) * LDA_ + (hi << 3)];
    const v8h a1 = *(const v8h*)&As[(tm + hl) * LDA_ + (hi << 3) + 16];
    const v16h af = __builtin_shufflevector(a0, a1, 0, 1, 2, 3, 4, 5, 6, 7,
                                            8, 9, 10, 11, 12, 13, 14, 15);
    const v8h b0 = *(const v8h*)&BsT[(tn + hl) * LDA_ + (hi << 3)];
    const v8h b1 = *(const v8h*)&BsT[(tn + hl) * LDA_ + (hi << 3) + 16];
    const v16h bf0 = __builtin_shufflevector(b0, b1, 0, 1, 2, 3, 4, 5, 6, 7,
                                             8, 9, 10, 11, 12, 13, 14, 15);
    const v8h b2 = *(const v8h*)&BsT[(tn + 16 + hl) * LDA_ + (hi << 3)];
    const v8h b3 = *(const v8h*)&BsT[(tn + 16 + hl) * LDA_ + (hi << 3) + 16];
    const v16h bf1 = __builtin_shufflevector(b2, b3, 0, 1, 2, 3, 4, 5, 6, 7,
                                             8, 9, 10, 11, 12, 13, 14, 15);
    acc0 = __builtin_amdgcn_wmma_f32_16x16x32_f16(false, af, false, bf0,
                                                  (short)0, acc0, false, false);
    acc1 = __builtin_amdgcn_wmma_f32_16x16x32_f16(false, af, false, bf1,
                                                  (short)0, acc1, false, false);
    __syncthreads();
  }
  // epilogue: bias + activation, guarded rows
#pragma unroll
  for (int t = 0; t < 2; ++t) {
    const v8f acc = t ? acc1 : acc0;
    const int col = colBase + tn + (t << 4) + hl;
    const float bv = bias ? bias[col] : 0.0f;
#pragma unroll
    for (int r = 0; r < 8; ++r) {
      int row = rowBase + tm + r + (hi << 3);  // VGPR r: rows r / r+8
      if (row < M) {
        float v = acc[r] + bv;
        if (ACT == 1) v = fmaxf(v, 0.0f);
        C[(long long)row * Nn + col] = v;
      }
    }
  }
}

// =====================================================================
// LayerNorm over last dim (C=64 or 192), one wave32 per row.
// bmode: 0 -> ln(alpha*a); 1 -> ln(a+b); 2 -> ln(a + b[row % bper])
// =====================================================================
__global__ __launch_bounds__(256) void ln_kernel(
    const float* __restrict__ a, const float* __restrict__ b, int bmode, float alpha,
    const float* __restrict__ g, const float* __restrict__ be,
    float* __restrict__ out, int rows, int Cc, int bper) {
  int wid = (int)((blockIdx.x * 256 + threadIdx.x) >> 5);
  int lane = threadIdx.x & 31;
  if (wid >= rows) return;
  const float* ar = a + (long long)wid * Cc;
  const float* br = nullptr;
  if (bmode == 1) br = b + (long long)wid * Cc;
  else if (bmode == 2) br = b + (long long)(wid % bper) * Cc;
  float vals[6];
  const int nch = Cc >> 5;
  float s = 0.0f;
  for (int i = 0; i < nch; ++i) {
    int c = lane + (i << 5);
    float v = ar[c];
    if (br) v += br[c];
    v *= alpha;
    vals[i] = v;
    s += v;
  }
  for (int o = 16; o > 0; o >>= 1) s += __shfl_xor(s, o, 32);
  float mean = s / (float)Cc;
  float s2 = 0.0f;
  for (int i = 0; i < nch; ++i) { float d = vals[i] - mean; s2 += d * d; }
  for (int o = 16; o > 0; o >>= 1) s2 += __shfl_xor(s2, o, 32);
  float inv = rsqrtf(s2 / (float)Cc + 1e-5f);
  float* orow = out + (long long)wid * Cc;
  for (int i = 0; i < nch; ++i) {
    int c = lane + (i << 5);
    orow[c] = (vals[i] - mean) * inv * g[c] + be[c];
  }
}

// =====================================================================
// Spatial attention over N=307 (hd=16, heads=4).  One block per (b,t,h),
// K/V tiles resident in LDS, online softmax per query row.
// =====================================================================
__global__ __launch_bounds__(320) void spatial_attn(
    const float* __restrict__ Q, const float* __restrict__ K,
    const float* __restrict__ V, float* __restrict__ ctx) {
  __shared__ float Ks[N_ * 16];
  __shared__ float Vs[N_ * 16];
  const int h = blockIdx.x & 3;
  const int t = (blockIdx.x >> 2) % T_;
  const int b = blockIdx.x / (T_ * H_);
  for (int i = threadIdx.x; i < N_ * 16; i += 320) {
    int m = i >> 4, d = i & 15;
    long long idx = ((((long long)b * N_ + m) * T_) + t) * C_ + h * 16 + d;
    Ks[i] = K[idx];
    Vs[i] = V[idx];
  }
  __syncthreads();
  const int n = threadIdx.x;
  if (n >= N_) return;
  long long qi = ((((long long)b * N_ + n) * T_) + t) * C_ + h * 16;
  float q[16];
#pragma unroll
  for (int d = 0; d < 16; ++d) q[d] = Q[qi + d];
  float mx = -1e30f, l = 0.0f, acc[16];
#pragma unroll
  for (int d = 0; d < 16; ++d) acc[d] = 0.0f;
  for (int m = 0; m < N_; ++m) {
    float s = 0.0f;
#pragma unroll
    for (int d = 0; d < 16; ++d) s += q[d] * Ks[m * 16 + d];
    s *= 0.25f;  // 1/sqrt(16)
    if (s > mx) {
      float r = __expf(mx - s);
      l *= r;
#pragma unroll
      for (int d = 0; d < 16; ++d) acc[d] *= r;
      mx = s;
    }
    float e = __expf(s - mx);
    l += e;
#pragma unroll
    for (int d = 0; d < 16; ++d) acc[d] += e * Vs[m * 16 + d];
  }
  float inv = 1.0f / l;
#pragma unroll
  for (int d = 0; d < 16; ++d) ctx[qi + d] = acc[d] * inv;
}

// =====================================================================
// Temporal attention over T=12; one thread per (b,n,h,s).
// =====================================================================
template <int CC, int HD>
__global__ __launch_bounds__(256) void temporal_attn(
    const float* __restrict__ Q, const float* __restrict__ K,
    const float* __restrict__ V, float* __restrict__ ctx) {
  constexpr int HH = CC / HD;
  long long gid = (long long)blockIdx.x * 256 + threadIdx.x;
  long long total = (long long)B_ * N_ * HH * T_;
  if (gid >= total) return;
  int s = (int)(gid % T_);
  long long t1 = gid / T_;
  int h = (int)(t1 % HH);
  long long bn = t1 / HH;
  const float scale = rsqrtf((float)HD);
  long long base = bn * (long long)T_ * CC + h * HD;
  float q[HD];
#pragma unroll
  for (int d = 0; d < HD; ++d) q[d] = Q[base + (long long)s * CC + d];
  float sc[T_];
  float mx = -1e30f;
  for (int t = 0; t < T_; ++t) {
    float dot = 0.0f;
#pragma unroll
    for (int d = 0; d < HD; ++d) dot += q[d] * K[base + (long long)t * CC + d];
    dot *= scale;
    sc[t] = dot;
    mx = fmaxf(mx, dot);
  }
  float l = 0.0f;
  for (int t = 0; t < T_; ++t) { sc[t] = __expf(sc[t] - mx); l += sc[t]; }
  float inv = 1.0f / l;
#pragma unroll
  for (int d = 0; d < HD; ++d) {
    float acc = 0.0f;
    for (int t = 0; t < T_; ++t) acc += sc[t] * V[base + (long long)t * CC + d];
    ctx[base + (long long)s * CC + d] = acc * inv;
  }
}

// ---- adjacency normalization ----
__global__ void rowsum_kernel(const float* __restrict__ adj, float* __restrict__ deg) {
  int i = blockIdx.x * blockDim.x + threadIdx.x;
  if (i >= N_) return;
  float s = 0.0f;
  for (int j = 0; j < N_; ++j) s += adj[(long long)i * N_ + j];
  deg[i] = s;
}
__global__ void anorm_kernel(const float* __restrict__ adj, const float* __restrict__ deg,
                             float* __restrict__ an) {
  int idx = blockIdx.x * blockDim.x + threadIdx.x;
  if (idx >= AN_) return;
  int i = idx / N_, j = idx % N_;
  float di = deg[i] > 0.0f ? rsqrtf(deg[i]) : 0.0f;
  float dj = deg[j] > 0.0f ? rsqrtf(deg[j]) : 0.0f;
  an[idx] = di * adj[idx] * dj;
}

// ---- gated tanh*sigmoid temporal conv + max-pool over time ----
// pooled layout: [B, 12(out ch), N]
__global__ void gated_conv(const float* __restrict__ x,
                           const float* __restrict__ w1, const float* __restrict__ b1,
                           const float* __restrict__ w2, const float* __restrict__ b2,
                           int kt, float* __restrict__ pooled) {
  int gid = blockIdx.x * blockDim.x + threadIdx.x;
  int total = B_ * T_ * N_;  // 12 output channels == T_
  if (gid >= total) return;
  int n = gid % N_;
  int tmp = gid / N_;
  int oc = tmp % T_;
  int b = tmp / T_;
  int tlen = T_ - kt + 1;
  float best = -1e30f;
  for (int to = 0; to < tlen; ++to) {
    float s1 = b1[oc], s2 = b2[oc];
    for (int k = 0; k < kt; ++k) {
      const float* xr = &x[(((long long)b * N_ + n) * T_ + to + k) * C_];
      const float* w1r = &w1[((long long)oc * C_) * kt + k];
      const float* w2r = &w2[((long long)oc * C_) * kt + k];
      for (int c = 0; c < C_; ++c) {
        float xv = xr[c];
        s1 += xv * w1r[(long long)c * kt];
        s2 += xv * w2r[(long long)c * kt];
      }
    }
    float v = tanhf(s1) * sigm(s2);
    best = fmaxf(best, v);
  }
  pooled[gid] = best;
}

// out[b,n,t,c] = pooled[b,t,n] * cw[c] + cb[c]
__global__ void expand_pooled(const float* __restrict__ pooled,
                              const float* __restrict__ cw, const float* __restrict__ cb,
                              float* __restrict__ out) {
  long long i = (long long)blockIdx.x * blockDim.x + threadIdx.x;
  if (i >= SZ_) return;
  int c = (int)(i & (C_ - 1));
  long long row = i >> 6;
  int t = (int)(row % T_);
  long long bn = row / T_;
  int n = (int)(bn % N_);
  int b = (int)(bn / N_);
  out[i] = pooled[((long long)b * T_ + t) * N_ + n] * cw[c] + cb[c];
}

// out = g*u + (1-g)*v,  g = sigmoid(t1+t2)
__global__ void gate_fuse(const float* __restrict__ t1, const float* __restrict__ t2,
                          const float* __restrict__ u, const float* __restrict__ v,
                          float* __restrict__ out, long long nelem) {
  long long i = (long long)blockIdx.x * blockDim.x + threadIdx.x;
  if (i >= nelem) return;
  float g = sigm(t1[i] + t2[i]);
  out[i] = g * u[i] + (1.0f - g) * v[i];
}

// feats[row, 0:64]=s ; [64:128]=t ; [128:192]=adp[row % NT]
__global__ void concat_feats(const float* __restrict__ s_, const float* __restrict__ t_,
                             const float* __restrict__ adp, float* __restrict__ feats) {
  long long i = (long long)blockIdx.x * blockDim.x + threadIdx.x;
  if (i >= SZ3_) return;
  int c = (int)(i % D3_);
  long long row = i / D3_;
  float v;
  if (c < C_)            v = s_[row * C_ + c];
  else if (c < 2 * C_)   v = t_[row * C_ + (c - C_)];
  else                   v = adp[(row % NT_) * C_ + (c - 2 * C_)];
  feats[i] = v;
}

// =====================================================================
// Host orchestration
// =====================================================================
static inline void launch_gemm(hipStream_t s, const float* A, const float* Wt,
                               const float* bias, float* C, int M, int Nn, int K,
                               int batch, long long sA, long long sW, long long sC,
                               int act) {
  dim3 grid(Nn / 64, (M + 63) / 64, batch), block(256);
  if (act)
    gemm_wmma<1><<<grid, block, 0, s>>>(A, Wt, bias, C, M, Nn, K, sA, sW, sC);
  else
    gemm_wmma<0><<<grid, block, 0, s>>>(A, Wt, bias, C, M, Nn, K, sA, sW, sC);
}
static inline void launch_ln(hipStream_t s, const float* a, const float* b, int bmode,
                             float alpha, const float* g, const float* be, float* out,
                             int rows, int Cc, int bper) {
  int blocks = (rows + 7) / 8;  // 8 wave32 rows per 256-thread block
  ln_kernel<<<blocks, 256, 0, s>>>(a, b, bmode, alpha, g, be, out, rows, Cc, bper);
}

extern "C" void kernel_launch(void* const* d_in, const int* in_sizes, int n_in,
                              void* d_out, int out_size, void* d_ws, size_t ws_size,
                              hipStream_t stream) {
  (void)in_sizes; (void)n_in; (void)out_size; (void)ws_size;
  auto in = [&](int i) { return (const float*)d_in[i]; };
  const float* x   = in(IN_X);
  const float* adj = in(IN_ADJ);
  float* outp = (float*)d_out;

  // ---- workspace bump allocation (floats) ----
  float* w = (float*)d_ws;
  size_t off = 0;
  auto take = [&](long long nf) {
    float* p = w + off;
    off += (size_t)((nf + 255) & ~255LL);
    return p;
  };
  float* anorm = take(AN_);
  float* deg   = take(N_);
  float* e1b   = take(AROW_);
  float* xeb   = take(AROW_);
  float* poolg = take((long long)B_ * T_ * N_);
  float* pooll = take((long long)B_ * T_ * N_);
  float* P[9];
  for (int i = 0; i < 9; ++i) P[i] = take(SZ_);
  float* F[5];
  for (int i = 0; i < 5; ++i) F[i] = take(SZ3_);
  float* MIDB = take(MIDN_);

  auto GSZ = [](long long n) { return (unsigned)((n + 255) / 256); };

  // ================= AGCN =================
  rowsum_kernel<<<GSZ(N_), 256, 0, stream>>>(adj, deg);
  anorm_kernel<<<GSZ(AN_), 256, 0, stream>>>(adj, deg, anorm);
  // G1[b] = A_norm @ x[b]  ([307,307]@[307,768], batched over B)
  launch_gemm(stream, anorm, x, nullptr, P[0], N_, TC_, N_, B_, 0, AROW_, AROW_, 0);
  // x_g = relu(G1 @ ff1 + b)
  launch_gemm(stream, P[0], in(FF1_W), in(FF1_B), P[1], R_, C_, C_, 1, 0, 0, 0, 1);
  // x_e = (relu(adj@E1+b)) @ E2 + b
  launch_gemm(stream, adj, in(E1_W), in(E1_B), e1b, N_, TC_, N_, 1, 0, 0, 0, 1);
  launch_gemm(stream, e1b, in(E2_W), in(E2_B), xeb, N_, TC_, TC_, 1, 0, 0, 0, 0);
  // q = LN(x + x_e_broadcast, n5)
  launch_ln(stream, x, xeb, 2, 1.0f, in(N5_G), in(N5_B), P[2], R_, C_, NT_);
  // spatial MHA #1
  launch_gemm(stream, P[2], in(A1_WQ), nullptr, P[3], R_, C_, C_, 1, 0, 0, 0, 0);
  launch_gemm(stream, P[2], in(A1_WK), nullptr, P[4], R_, C_, C_, 1, 0, 0, 0, 0);
  launch_gemm(stream, P[2], in(A1_WV), nullptr, P[5], R_, C_, C_, 1, 0, 0, 0, 0);
  spatial_attn<<<B_ * T_ * H_, 320, 0, stream>>>(P[3], P[4], P[5], P[6]);
  launch_gemm(stream, P[6], in(A1_OW), in(A1_OB), P[3], R_, C_, C_, 1, 0, 0, 0, 0);
  launch_ln(stream, P[3], P[2], 1, 1.0f, in(N1_G), in(N1_B), P[4], R_, C_, 0);  // h
  // u_s = LN(ffn(h)+h, n2)
  launch_gemm(stream, P[4], in(FF_L1W), in(FF_L1B), MIDB, R_, 256, C_, 1, 0, 0, 0, 1);
  launch_gemm(stream, MIDB, in(FF_L2W), in(FF_L2B), P[5], R_, C_, 256, 1, 0, 0, 0, 0);
  launch_ln(stream, P[5], P[4], 1, 1.0f, in(N2_G), in(N2_B), P[6], R_, C_, 0);  // u_s
  // xg = LN(x + x_g, n6)
  launch_ln(stream, x, P[1], 1, 1.0f, in(N6_G), in(N6_B), P[3], R_, C_, 0);
  // spatial MHA #2
  launch_gemm(stream, P[3], in(AT2_WQ), nullptr, P[1], R_, C_, C_, 1, 0, 0, 0, 0);
  launch_gemm(stream, P[3], in(AT2_WK), nullptr, P[4], R_, C_, C_, 1, 0, 0, 0, 0);
  launch_gemm(stream, P[3], in(AT2_WV), nullptr, P[5], R_, C_, C_, 1, 0, 0, 0, 0);
  spatial_attn<<<B_ * T_ * H_, 320, 0, stream>>>(P[1], P[4], P[5], P[7]);
  launch_gemm(stream, P[7], in(AT2_OW), in(AT2_OB), P[1], R_, C_, C_, 1, 0, 0, 0, 0);
  launch_ln(stream, P[1], P[3], 1, 1.0f, in(N3_G), in(N3_B), P[4], R_, C_, 0);  // y
  launch_gemm(stream, P[4], in(F1N_L1W), in(F1N_L1B), MIDB, R_, 256, C_, 1, 0, 0, 0, 1);
  launch_gemm(stream, MIDB, in(F1N_L2W), in(F1N_L2B), P[5], R_, C_, 256, 1, 0, 0, 0, 0);
  launch_ln(stream, P[5], P[4], 1, 1.0f, in(N4_G), in(N4_B), P[7], R_, C_, 0);  // xg2
  // gate: sigmoid(us@fs + xg2@fg)
  launch_gemm(stream, P[6], in(AFS_W), in(AFS_B), P[1], R_, C_, C_, 1, 0, 0, 0, 0);
  launch_gemm(stream, P[7], in(AFG_W), in(AFG_B), P[4], R_, C_, C_, 1, 0, 0, 0, 0);
  gate_fuse<<<GSZ(SZ_), 256, 0, stream>>>(P[1], P[4], P[6], P[7], P[5], SZ_);
  // output_s = LN(agcn + x, norm1)  (kept in P[8] until concat)
  launch_ln(stream, P[5], x, 1, 1.0f, in(NORM1_G), in(NORM1_B), P[8], R_, C_, 0);

  // ================= temporal branches =================
  // global gated conv (kt = T)
  gated_conv<<<GSZ((long long)B_ * T_ * N_), 256, 0, stream>>>(
      x, in(GG1_W), in(GG1_B), in(GG2_W), in(GG2_B), T_, poolg);
  expand_pooled<<<GSZ(SZ_), 256, 0, stream>>>(poolg, in(C4_W), in(C4_B), P[0]);
  launch_ln(stream, P[0], x, 1, 1.0f, in(NORM3_G), in(NORM3_B), P[1], R_, C_, 0);
  launch_gemm(stream, P[1], in(TG_WQ), nullptr, P[2], R_, C_, C_, 1, 0, 0, 0, 0);
  launch_gemm(stream, P[1], in(TG_WK), nullptr, P[3], R_, C_, C_, 1, 0, 0, 0, 0);
  launch_gemm(stream, P[1], in(TG_WV), nullptr, P[4], R_, C_, C_, 1, 0, 0, 0, 0);
  temporal_attn<C_, 16><<<GSZ((long long)B_ * N_ * H_ * T_), 256, 0, stream>>>(
      P[2], P[3], P[4], P[5]);
  launch_gemm(stream, P[5], in(TG_OW), in(TG_OB), P[6], R_, C_, C_, 1, 0, 0, 0, 0); // out_g
  // local gated conv (kt = 4)
  gated_conv<<<GSZ((long long)B_ * T_ * N_), 256, 0, stream>>>(
      x, in(LG1_W), in(LG1_B), in(LG2_W), in(LG2_B), 4, pooll);
  expand_pooled<<<GSZ(SZ_), 256, 0, stream>>>(pooll, in(C5_W), in(C5_B), P[0]);
  launch_ln(stream, P[0], x, 1, 1.0f, in(NORM4_G), in(NORM4_B), P[1], R_, C_, 0);
  launch_gemm(stream, P[1], in(TL_WQ), nullptr, P[2], R_, C_, C_, 1, 0, 0, 0, 0);
  launch_gemm(stream, P[1], in(TL_WK), nullptr, P[3], R_, C_, C_, 1, 0, 0, 0, 0);
  launch_gemm(stream, P[1], in(TL_WV), nullptr, P[4], R_, C_, C_, 1, 0, 0, 0, 0);
  temporal_attn<C_, 16><<<GSZ((long long)B_ * N_ * H_ * T_), 256, 0, stream>>>(
      P[2], P[3], P[4], P[5]);
  launch_gemm(stream, P[5], in(TL_OW), in(TL_OB), P[7], R_, C_, C_, 1, 0, 0, 0, 0); // out_l
  // g = sigmoid(out_l@fs + out_g@fg)  rows = B*N, dim = 768
  launch_gemm(stream, P[7], in(FS_W), in(FS_B), P[0], B_ * N_, TC_, TC_, 1, 0, 0, 0, 0);
  launch_gemm(stream, P[6], in(FG_W), in(FG_B), P[1], B_ * N_, TC_, TC_, 1, 0, 0, 0, 0);
  gate_fuse<<<GSZ(SZ_), 256, 0, stream>>>(P[0], P[1], P[7], P[6], P[2], SZ_);
  // output_t = LN(2 * output_t, norm2)
  launch_ln(stream, P[2], nullptr, 0, 2.0f, in(NORM2_G), in(NORM2_B), P[3], R_, C_, 0);

  // ================= STGA =================
  concat_feats<<<GSZ(SZ3_), 256, 0, stream>>>(P[8], P[3], in(ADP), F[0]);
  launch_gemm(stream, F[0], in(SA_WQ), nullptr, F[1], R_, D3_, D3_, 1, 0, 0, 0, 0);
  launch_gemm(stream, F[0], in(SA_WK), nullptr, F[2], R_, D3_, D3_, 1, 0, 0, 0, 0);
  launch_gemm(stream, F[0], in(SA_WV), nullptr, F[3], R_, D3_, D3_, 1, 0, 0, 0, 0);
  temporal_attn<D3_, 48><<<GSZ((long long)B_ * N_ * H_ * T_), 256, 0, stream>>>(
      F[1], F[2], F[3], F[4]);
  launch_gemm(stream, F[4], in(SA_OW), in(SA_OB), F[1], R_, D3_, D3_, 1, 0, 0, 0, 0);
  launch_ln(stream, F[0], F[1], 1, 1.0f, in(SL1_G), in(SL1_B), F[2], R_, D3_, 0);  // h
  launch_gemm(stream, F[2], in(SF_L1W), in(SF_L1B), MIDB, R_, 128, D3_, 1, 0, 0, 0, 1);
  launch_gemm(stream, MIDB, in(SF_L2W), in(SF_L2B), F[3], R_, D3_, 128, 1, 0, 0, 0, 0);
  launch_ln(stream, F[3], F[2], 1, 1.0f, in(SL2_G), in(SL2_B), F[1], R_, D3_, 0);
  // final projection -> d_out [B,N,T,64]
  launch_gemm(stream, F[1], in(OP_W), in(OP_B), outp, R_, C_, D3_, 1, 0, 0, 0, 0);
}
